// MinamoTopoModel_22488448762459
// MI455X (gfx1250) — matrix operand order repaired and checked by reading, not compiled, not run on hardware
//
#include <hip/hip_runtime.h>
#include <hip/hip_bf16.h>
#include <limits>

typedef __bf16 bf16_t;
typedef bf16_t v16bf __attribute__((ext_vector_type(16)));
typedef bf16_t v8bf  __attribute__((ext_vector_type(8)));
typedef float  v8f   __attribute__((ext_vector_type(8)));

#define BM 128
#define BN 128
#define BK 32

// ---------------------------------------------------------------- utilities

__device__ __forceinline__ void atomicMaxFloat(float* addr, float val) {
  // ordered-int trick; works for mixed signs with -inf init
  if (val >= 0.0f) atomicMax((int*)addr, __float_as_int(val));
  else             atomicMin((unsigned int*)addr, __float_as_uint(val));
}

__device__ __forceinline__ float block_sum(float v, float* red) {
  int lane = threadIdx.x & 31, w = threadIdx.x >> 5;
  for (int o = 16; o; o >>= 1) v += __shfl_xor(v, o, 32);
  __syncthreads();
  if (lane == 0) red[w] = v;
  __syncthreads();
  float t = 0.0f;
  int nw = blockDim.x >> 5;
  for (int i = 0; i < nw; ++i) t += red[i];
  return t;
}

__device__ __forceinline__ v16bf cat8(v8bf lo, v8bf hi) {
  return __builtin_shufflevector(lo, hi, 0,1,2,3,4,5,6,7,8,9,10,11,12,13,14,15);
}

// ---------------------------------------------------------------- fill / convert

__global__ void fill_f32(float* __restrict__ p, float v, long long n) {
  long long i = (long long)blockIdx.x * blockDim.x + threadIdx.x;
  long long st = (long long)gridDim.x * blockDim.x;
  for (; i < n; i += st) p[i] = v;
}

__global__ void cvt_f32_bf16(const float* __restrict__ in, bf16_t* __restrict__ out, long long n) {
  long long i = (long long)blockIdx.x * blockDim.x + threadIdx.x;
  long long st = (long long)gridDim.x * blockDim.x;
  for (; i < n; i += st) out[i] = (bf16_t)in[i];
}

// W: [K, Nn] f32 row-major  ->  Wt: [Nn, K] bf16 row-major (B^T for WMMA B frags)
__global__ void transpose_w_bf16(const float* __restrict__ W, bf16_t* __restrict__ Wt,
                                 int K, int Nn) {
  long long total = (long long)K * Nn;
  long long i = (long long)blockIdx.x * blockDim.x + threadIdx.x;
  long long st = (long long)gridDim.x * blockDim.x;
  for (; i < total; i += st) {
    int n = (int)(i / K);
    int k = (int)(i % K);
    Wt[i] = (bf16_t)W[(long long)k * Nn + n];
  }
}

// ---------------------------------------------------------------- projection x@Wp+bp

__global__ __launch_bounds__(128)
void proj_kernel(const float* __restrict__ x, const float* __restrict__ Wp,
                 const float* __restrict__ bp, float* __restrict__ out, int N) {
  __shared__ float xs[32];
  int n = blockIdx.x;
  int j = threadIdx.x;
  if (j < 32) xs[j] = x[(size_t)n * 32 + j];
  __syncthreads();
  float acc = bp[j];
  #pragma unroll
  for (int k = 0; k < 32; ++k) acc += xs[k] * Wp[k * 128 + j];
  out[(size_t)n * 128 + j] = acc;
}

// ---------------------------------------------------------------- WMMA GEMM (bf16 -> f32)
// A: [M,K] bf16 row-major, Bt: [Nn,K] bf16 row-major, C: [M,Nn] f32.

__global__ __launch_bounds__(256)
void gemm_bf16_wmma(const bf16_t* __restrict__ A, const bf16_t* __restrict__ Bt,
                    float* __restrict__ C, int M, int Nn, int K) {
  __shared__ __align__(16) bf16_t As[BM][BK];
  __shared__ __align__(16) bf16_t Bs[BN][BK];

  const int tid  = threadIdx.x;
  const int lane = tid & 31;
  const int wave = tid >> 5;
  const int bm = blockIdx.x * BM;
  const int bn = blockIdx.y * BN;
  // 8 waves: 4 in M x 2 in N, each wave computes 32(M) x 64(N)
  const int wm = (wave >> 1) * 32;
  const int wn = (wave & 1) * 64;

  v8f acc[2][4];
  #pragma unroll
  for (int i = 0; i < 2; ++i)
    #pragma unroll
    for (int j = 0; j < 4; ++j)
      #pragma unroll
      for (int r = 0; r < 8; ++r) acc[i][j][r] = 0.0f;

  const int lrow = tid >> 1;          // 0..127
  const int lcol = (tid & 1) * 16;    // 0 or 16

  int arow = bm + lrow; if (arow >= M)  arow = M - 1;   // clamp; rows >= M never stored
  int brow = bn + lrow; if (brow >= Nn) brow = Nn - 1;
  const bf16_t* aptr = A  + (size_t)arow * K + lcol;
  const bf16_t* bptr = Bt + (size_t)brow * K + lcol;

  const int nlane   = lane & 15;
  const int halfsel = lane >> 4;
  const int kA = halfsel * 8;    // A frag: chunks at kA and kA+16
  const int kB = halfsel * 8;    // B frag: chunks at kB and kB+16

  for (int k0 = 0; k0 < K; k0 += BK) {
    v8bf a0 = *(const v8bf*)(aptr + k0);
    v8bf a1 = *(const v8bf*)(aptr + k0 + 8);
    v8bf b0 = *(const v8bf*)(bptr + k0);
    v8bf b1 = *(const v8bf*)(bptr + k0 + 8);
    if (k0 + BK < K) {                                  // gfx1250 global_prefetch_b8
      __builtin_prefetch(aptr + k0 + BK, 0, 1);
      __builtin_prefetch(bptr + k0 + BK, 0, 1);
    }
    __syncthreads();
    *(v8bf*)&As[lrow][lcol]     = a0;
    *(v8bf*)&As[lrow][lcol + 8] = a1;
    *(v8bf*)&Bs[lrow][lcol]     = b0;
    *(v8bf*)&Bs[lrow][lcol + 8] = b1;
    __syncthreads();

    // Load ALL fragments (12x ds_load_b128) before any WMMA.
    v16bf afrag[2];
    v16bf bfrag[4];
    #pragma unroll
    for (int i = 0; i < 2; ++i) {
      const bf16_t* s = &As[wm + i * 16 + nlane][0];
      afrag[i] = cat8(*(const v8bf*)(s + kA), *(const v8bf*)(s + kA + 16));
    }
    #pragma unroll
    for (int j = 0; j < 4; ++j) {
      const bf16_t* s = &Bs[wn + j * 16 + nlane][kB];
      bfrag[j] = cat8(*(const v8bf*)s, *(const v8bf*)(s + 16));
    }
#if __has_builtin(__builtin_amdgcn_sched_barrier)
    // Fence the scheduler: keep every fragment in distinct VGPRs so the 8
    // WMMAs below issue back-to-back after a single s_wait_dscnt, instead of
    // ping-ponging B frags through one register octet with waits in between.
    __builtin_amdgcn_sched_barrier(0);
#endif
    #pragma unroll
    for (int j = 0; j < 4; ++j)
      #pragma unroll
      for (int i = 0; i < 2; ++i)
        acc[i][j] = __builtin_amdgcn_wmma_f32_16x16x32_bf16(
            false, afrag[i], false, bfrag[j], (short)0, acc[i][j], false, false);
  }

  #pragma unroll
  for (int i = 0; i < 2; ++i)
    #pragma unroll
    for (int j = 0; j < 4; ++j) {
      int col = bn + wn + j * 16 + nlane;
      #pragma unroll
      for (int r = 0; r < 8; ++r) {
        int row = bm + wm + i * 16 + halfsel * 8 + r;
        if (row < M) C[(size_t)row * Nn + col] = acc[i][j][r];
      }
    }
}

// ---------------------------------------------------------------- attention pieces

// one wave per (node, head): al_s = sum_c h*a_s, al_d = sum_c h*a_d
__global__ __launch_bounds__(256)
void att_coeff_kernel(const float* __restrict__ h, const float* __restrict__ a_s,
                      const float* __restrict__ a_d, float* __restrict__ als,
                      float* __restrict__ ald, int N, int H, int C) {
  int wave = threadIdx.x >> 5, lane = threadIdx.x & 31;
  long long pair = (long long)blockIdx.x * 8 + wave;
  if (pair >= (long long)N * H) return;
  int node = (int)(pair / H), head = (int)(pair % H);
  const float* hrow = h + (size_t)node * H * C + (size_t)head * C;
  float ss = 0.0f, sd = 0.0f;
  for (int c = lane; c < C; c += 32) {
    float hv = hrow[c];
    ss += hv * a_s[head * C + c];
    sd += hv * a_d[head * C + c];
  }
  for (int o = 16; o; o >>= 1) { ss += __shfl_xor(ss, o, 32); sd += __shfl_xor(sd, o, 32); }
  if (lane == 0) { als[(size_t)node * H + head] = ss; ald[(size_t)node * H + head] = sd; }
}

__global__ void edge_max_kernel(const int* __restrict__ ei, const float* __restrict__ als,
                                const float* __restrict__ ald, float* __restrict__ mbuf,
                                int E, int N, int H) {
  long long total = (long long)(E + N) * H;
  long long i = (long long)blockIdx.x * blockDim.x + threadIdx.x;
  long long st = (long long)gridDim.x * blockDim.x;
  for (; i < total; i += st) {
    int e = (int)(i / H), h = (int)(i % H);
    int src = e < E ? ei[e]     : e - E;
    int dst = e < E ? ei[E + e] : e - E;
    float sc = als[(size_t)src * H + h] + ald[(size_t)dst * H + h];
    sc = sc > 0.0f ? sc : 0.2f * sc;                 // leaky_relu(0.2)
    atomicMaxFloat(&mbuf[(size_t)dst * H + h], sc);
  }
}

__global__ void edge_sum_kernel(const int* __restrict__ ei, const float* __restrict__ als,
                                const float* __restrict__ ald, const float* __restrict__ mbuf,
                                float* __restrict__ pbuf, float* __restrict__ sbuf,
                                int E, int N, int H) {
  long long total = (long long)(E + N) * H;
  long long i = (long long)blockIdx.x * blockDim.x + threadIdx.x;
  long long st = (long long)gridDim.x * blockDim.x;
  for (; i < total; i += st) {
    int e = (int)(i / H), h = (int)(i % H);
    int src = e < E ? ei[e]     : e - E;
    int dst = e < E ? ei[E + e] : e - E;
    float sc = als[(size_t)src * H + h] + ald[(size_t)dst * H + h];
    sc = sc > 0.0f ? sc : 0.2f * sc;
    float p = __expf(sc - mbuf[(size_t)dst * H + h]);
    pbuf[i] = p;
    atomicAdd(&sbuf[(size_t)dst * H + h], p);
  }
}

// one block per edge: out[dst, :] += alpha[head] * h[src, :]
__global__ __launch_bounds__(256)
void scatter_kernel(const int* __restrict__ ei, const float* __restrict__ hpre,
                    const float* __restrict__ pbuf, const float* __restrict__ sbuf,
                    float* __restrict__ out, int E, int N, int H, int C) {
  __shared__ int ssrc, sdst;
  int e = blockIdx.x;
  if (threadIdx.x == 0) {
    ssrc = e < E ? ei[e]     : e - E;
    sdst = e < E ? ei[E + e] : e - E;
  }
  __syncthreads();
  int src = ssrc, dst = sdst;
  int D = H * C;
  for (int c = threadIdx.x; c < D; c += blockDim.x) {
    int head = c / C;
    float alpha = pbuf[(size_t)e * H + head] / (sbuf[(size_t)dst * H + head] + 1e-16f);
    atomicAdd(&out[(size_t)dst * D + c], alpha * hpre[(size_t)src * D + c]);
  }
}

// out = relu(layernorm(x + b) * g + be), one block per node, D multiple of 256, D<=2048
__global__ __launch_bounds__(256)
void ln_relu_kernel(const float* __restrict__ x, const float* __restrict__ b,
                    const float* __restrict__ g, const float* __restrict__ be,
                    float* __restrict__ out, int D) {
  __shared__ float red[8];
  int n = blockIdx.x;
  int nc = D >> 8;
  float v[8];
  float s = 0.0f;
  for (int i = 0; i < nc; ++i) {
    int c = i * 256 + threadIdx.x;
    float t = x[(size_t)n * D + c] + b[c];
    v[i] = t; s += t;
  }
  float mu = block_sum(s, red) / (float)D;
  float q = 0.0f;
  for (int i = 0; i < nc; ++i) { float d = v[i] - mu; q += d * d; }
  float var = block_sum(q, red) / (float)D;
  float rs = rsqrtf(var + 1e-5f);
  for (int i = 0; i < nc; ++i) {
    int c = i * 256 + threadIdx.x;
    float o = (v[i] - mu) * rs * g[c] + be[c];
    out[(size_t)n * D + c] = o > 0.0f ? o : 0.0f;
  }
}

// ---------------------------------------------------------------- pool + head

__global__ void pool_kernel(const float* __restrict__ h, const int* __restrict__ batch,
                            float* __restrict__ pooled, int N) {
  long long total = (long long)N * 512;
  long long i = (long long)blockIdx.x * blockDim.x + threadIdx.x;
  long long st = (long long)gridDim.x * blockDim.x;
  for (; i < total; i += st) {
    int n = (int)(i / 512), c = (int)(i % 512);
    atomicMaxFloat(&pooled[(size_t)batch[n] * 512 + c], h[(size_t)n * 512 + c]);
  }
}

__global__ __launch_bounds__(256)
void fc_norm_kernel(const float* __restrict__ pooled, const float* __restrict__ Wfc,
                    const float* __restrict__ bfc, float* __restrict__ out) {
  __shared__ float sp[512];
  __shared__ float red[8];
  int g = blockIdx.x;
  for (int k = threadIdx.x; k < 512; k += 256) sp[k] = pooled[(size_t)g * 512 + k];
  __syncthreads();
  int j0 = threadIdx.x, j1 = threadIdx.x + 256;
  float v0 = bfc[j0], v1 = bfc[j1];
  for (int k = 0; k < 512; ++k) {
    float pk = sp[k];
    v0 += pk * Wfc[(size_t)k * 512 + j0];
    v1 += pk * Wfc[(size_t)k * 512 + j1];
  }
  float ss = block_sum(v0 * v0 + v1 * v1, red);
  float inv = 1.0f / fmaxf(sqrtf(ss), 1e-12f);
  out[(size_t)g * 512 + j0] = v0 * inv;
  out[(size_t)g * 512 + j1] = v1 * inv;
}

// ---------------------------------------------------------------- host

extern "C" void kernel_launch(void* const* d_in, const int* in_sizes, int n_in,
                              void* d_out, int out_size, void* d_ws, size_t ws_size,
                              hipStream_t stream) {
  const float* x     = (const float*)d_in[0];
  const int*   ei    = (const int*)d_in[1];
  const int*   batch = (const int*)d_in[2];
  const float* Wp    = (const float*)d_in[3];
  const float* bp    = (const float*)d_in[4];
  const float* W[4]  = {(const float*)d_in[5],  (const float*)d_in[11], (const float*)d_in[17], (const float*)d_in[23]};
  const float* as_[4]= {(const float*)d_in[6],  (const float*)d_in[12], (const float*)d_in[18], (const float*)d_in[24]};
  const float* ad_[4]= {(const float*)d_in[7],  (const float*)d_in[13], (const float*)d_in[19], (const float*)d_in[25]};
  const float* b_[4] = {(const float*)d_in[8],  (const float*)d_in[14], (const float*)d_in[20], (const float*)d_in[26]};
  const float* g_[4] = {(const float*)d_in[9],  (const float*)d_in[15], (const float*)d_in[21], (const float*)d_in[27]};
  const float* be_[4]= {(const float*)d_in[10], (const float*)d_in[16], (const float*)d_in[22], (const float*)d_in[28]};
  const float* Wfc   = (const float*)d_in[29];
  const float* bfc   = (const float*)d_in[30];
  float* out = (float*)d_out;

  const int N = in_sizes[0] / 32;
  const int E = in_sizes[1] / 2;
  const int Etot = E + N;
  const float NEG = -3.3e38f;

  // workspace carve-up
  char* base = (char*)d_ws;
  size_t off = 0;
  auto carve = [&](size_t bytes) -> void* {
    void* p = base + off;
    off = (off + bytes + 255) & ~(size_t)255;
    return p;
  };
  float*  hA     = (float*) carve((size_t)N * 2048 * 4);
  float*  hB     = (float*) carve((size_t)N * 2048 * 4);
  bf16_t* hbf    = (bf16_t*)carve((size_t)N * 2048 * 2);
  bf16_t* wbf    = (bf16_t*)carve((size_t)2048 * 2048 * 2);
  float*  als    = (float*) carve((size_t)N * 8 * 4);
  float*  ald    = (float*) carve((size_t)N * 8 * 4);
  float*  mbuf   = (float*) carve((size_t)N * 8 * 4);
  float*  sbuf   = (float*) carve((size_t)N * 8 * 4);
  float*  pbuf   = (float*) carve((size_t)Etot * 8 * 4);
  float*  pooled = (float*) carve((size_t)16 * 512 * 4);

  auto gs = [](long long n) -> int { return (int)((n + 255) / 256); };

  // projection: [N,32] @ [32,128] + bp
  proj_kernel<<<N, 128, 0, stream>>>(x, Wp, bp, hA, N);

  const int Din[4]  = {128, 2048, 2048, 2048};
  const int Dout[4] = {2048, 2048, 2048, 512};
  const int Hh[4]   = {8, 8, 8, 1};
  const int Cc[4]   = {256, 256, 256, 512};

  for (int L = 0; L < 4; ++L) {
    long long nIn = (long long)N * Din[L];
    cvt_f32_bf16<<<gs(nIn), 256, 0, stream>>>(hA, hbf, nIn);
    long long nW = (long long)Din[L] * Dout[L];
    transpose_w_bf16<<<gs(nW), 256, 0, stream>>>(W[L], wbf, Din[L], Dout[L]);

    dim3 gg((N + BM - 1) / BM, Dout[L] / BN);
    gemm_bf16_wmma<<<gg, 256, 0, stream>>>(hbf, wbf, hA, N, Dout[L], Din[L]);  // hA = hpre

    long long pairs = (long long)N * Hh[L];
    att_coeff_kernel<<<(int)((pairs + 7) / 8), 256, 0, stream>>>(hA, as_[L], ad_[L],
                                                                 als, ald, N, Hh[L], Cc[L]);
    fill_f32<<<gs(pairs), 256, 0, stream>>>(mbuf, NEG, pairs);
    fill_f32<<<gs(pairs), 256, 0, stream>>>(sbuf, 0.0f, pairs);
    long long nOut = (long long)N * Dout[L];
    fill_f32<<<gs(nOut), 256, 0, stream>>>(hB, 0.0f, nOut);

    long long nes = (long long)Etot * Hh[L];
    edge_max_kernel<<<gs(nes), 256, 0, stream>>>(ei, als, ald, mbuf, E, N, Hh[L]);
    edge_sum_kernel<<<gs(nes), 256, 0, stream>>>(ei, als, ald, mbuf, pbuf, sbuf, E, N, Hh[L]);
    scatter_kernel<<<Etot, 256, 0, stream>>>(ei, hA, pbuf, sbuf, hB, E, N, Hh[L], Cc[L]);

    ln_relu_kernel<<<N, 256, 0, stream>>>(hB, b_[L], g_[L], be_[L], hA, Dout[L]);
  }

  fill_f32<<<gs(16 * 512), 256, 0, stream>>>(pooled, NEG, 16 * 512);
  pool_kernel<<<gs((long long)N * 512), 256, 0, stream>>>(hA, batch, pooled, N);
  fc_norm_kernel<<<16, 256, 0, stream>>>(pooled, Wfc, bfc, out);
}
